// Class_Attention_81621558493750
// MI455X (gfx1250) — compile-verified, matrix operand order checked by reading, not run on hardware
//
#include <hip/hip_runtime.h>

// ---------------------------------------------------------------------------
// Types for CDNA5 WMMA (wave32): D(f32 16x16) = A(bf16 16x32) x B(bf16 32x16) + C
// ---------------------------------------------------------------------------
typedef __bf16 bf16_t;
typedef __attribute__((ext_vector_type(16))) __bf16 v16bf;
typedef __attribute__((ext_vector_type(8)))  __bf16 v8bf;
typedef __attribute__((ext_vector_type(8)))  float  v8f;
typedef __attribute__((ext_vector_type(4)))  int    v4i;

#define DEV __device__ __forceinline__

// Problem constants (from reference)
#define BATCH 8
#define SEQ   1024
#define DIM   1024
#define HEADS 16
#define HDIM  64

// CDNA5 async global->LDS path (ASYNCcnt-tracked DMA), guarded so the file
// still compiles if the toolchain lacks the builtins.
#if defined(__AMDGCN__) &&                                        \
    __has_builtin(__builtin_amdgcn_global_load_async_to_lds_b128) && \
    __has_builtin(__builtin_amdgcn_s_wait_asynccnt)
#define HAVE_ASYNC 1
#else
#define HAVE_ASYNC 0
#endif

#define AS1 __attribute__((address_space(1)))
#define AS3 __attribute__((address_space(3)))

// ---------------------------------------------------------------------------
// Fragment loader (global, row-major): 16x32 bf16 tile.
//   lane L: row = L&15, kbase = (L>>4)*8; elems 0..7 -> K=kbase..+7,
//   elems 8..15 -> K=kbase+16..+23.   Two global_load_b128 per lane.
// ---------------------------------------------------------------------------
DEV v16bf load_frag(const bf16_t* __restrict__ p0, int ld, int lane) {
  const bf16_t* p = p0 + (size_t)(lane & 15) * (size_t)ld + ((lane >> 4) << 3);
  v8bf a = *reinterpret_cast<const v8bf*>(p);
  v8bf b = *reinterpret_cast<const v8bf*>(p + 16);
  return __builtin_shufflevector(a, b, 0, 1, 2, 3, 4, 5, 6, 7,
                                       8, 9, 10, 11, 12, 13, 14, 15);
}

// ---------------------------------------------------------------------------
// Fragment loader from a swizzled 128x32 bf16 LDS tile.
// Tile row stride = 32 bf16 (64B = 16 banks); the four 16B chunks of a row are
// stored at chunk index (c ^ ((row>>2)&3)) so that the 16-lane fragment read
// (rows r..r+15, fixed chunk) hits 16 distinct banks. 16B alignment kept.
// ---------------------------------------------------------------------------
DEV v16bf lds_frag(const bf16_t* tile, int row0, int lane) {
  const int r = row0 + (lane & 15);
  const int hi = lane >> 4;
  const int key = (r >> 2) & 3;
  const bf16_t* rp = tile + r * 32;
  v8bf a = *reinterpret_cast<const v8bf*>(rp + ((hi ^ key) << 3));
  v8bf b = *reinterpret_cast<const v8bf*>(rp + (((hi + 2) ^ key) << 3));
  return __builtin_shufflevector(a, b, 0, 1, 2, 3, 4, 5, 6, 7,
                                       8, 9, 10, 11, 12, 13, 14, 15);
}

DEV v8f wmma_bf16(v16bf a, v16bf b, v8f c) {
  return __builtin_amdgcn_wmma_f32_16x16x32_bf16(
      /*neg_a=*/false, a, /*neg_b=*/false, b,
      /*c_mod=*/(short)0, c, /*reuse_a=*/false, /*reuse_b=*/false);
}

// ---------------------------------------------------------------------------
// fp32 -> bf16 conversion
// ---------------------------------------------------------------------------
__global__ void cvt_f32_to_bf16(const float* __restrict__ in,
                                bf16_t* __restrict__ out, int n) {
  int i = blockIdx.x * blockDim.x + threadIdx.x;
  if (i < n) out[i] = (bf16_t)in[i];
}

// ---------------------------------------------------------------------------
// GEMM  y[M=8192, N=1024] = A[8192,1024] @ W[1024,1024]^T  (bf16 in, f32 acc)
// Block: 256 threads = 8 waves (4 in M x 2 in N). Block tile 128x128.
// Wave tile 32x64 -> 2x4 f32 accumulators, 8 WMMA per 32-wide K-step.
//
// Data movement (HAVE_ASYNC): 128x32 A-slab and 128x32 W-slab are DMA'd into
// double-buffered LDS with GLOBAL_LOAD_ASYNC_TO_LDS_B128 (4 ops/thread/tile),
// overlapped with WMMA on the other buffer; s_wait_asynccnt 4 + s_barrier
// separates producer/consumer per the ASYNCcnt model.
//
// MODE 0: out bf16 head-major [B,H,N,64]            (Q with scale, K)
// MODE 1: out bf16 head-major transposed [B,H,64,N] (V^T, vectorized store)
// MODE 2: out f32 row-major [M,1024] + bias         (final projection)
// ---------------------------------------------------------------------------
template <int MODE>
__global__ __launch_bounds__(256) void gemm_xwT(
    const bf16_t* __restrict__ A, const bf16_t* __restrict__ W,
    const float* __restrict__ bias, float scale, void* __restrict__ outp) {
  const int lane = threadIdx.x & 31;
  const int wave = threadIdx.x >> 5;
  const int wm = wave >> 1;           // 0..3
  const int wn = wave & 1;            // 0..1
  const int mblk = blockIdx.y * 128;
  const int nblk = blockIdx.x * 128;
  const int m0 = mblk + wm * 32;
  const int n0 = nblk + wn * 64;
  const int lo = lane & 15, hi = lane >> 4;

  v8f acc[2][4] = {};

#if HAVE_ASYNC
  __shared__ __align__(16) bf16_t lA[2][128 * 32];
  __shared__ __align__(16) bf16_t lW[2][128 * 32];

  // Each thread DMAs two 16B chunks of A and two of W per K-step.
  const int tid = threadIdx.x;
  int rowv[2], colv[2], swv[2];
#pragma unroll
  for (int i = 0; i < 2; ++i) {
    const int cid = i * 256 + tid;       // 512 chunks per 128x32 tile
    const int row = cid >> 2;
    const int c = cid & 3;
    rowv[i] = row;
    colv[i] = c << 3;                                    // bf16 column offset
    swv[i] = row * 32 + ((c ^ ((row >> 2) & 3)) << 3);   // swizzled LDS elems
  }

  auto issue_tile = [&](int kk, int buf) {
#pragma unroll
    for (int i = 0; i < 2; ++i) {
      __builtin_amdgcn_global_load_async_to_lds_b128(
          (AS1 v4i*)(A + (size_t)(mblk + rowv[i]) * DIM + kk + colv[i]),
          (AS3 v4i*)(&lA[buf][swv[i]]), 0, 0);
    }
#pragma unroll
    for (int i = 0; i < 2; ++i) {
      __builtin_amdgcn_global_load_async_to_lds_b128(
          (AS1 v4i*)(W + (size_t)(nblk + rowv[i]) * DIM + kk + colv[i]),
          (AS3 v4i*)(&lW[buf][swv[i]]), 0, 0);
    }
  };

  issue_tile(0, 0);

#pragma unroll 1
  for (int kk = 0; kk < DIM; kk += 32) {
    const int buf = (kk >> 5) & 1;
    if (kk + 32 < DIM) {
      issue_tile(kk + 32, buf ^ 1);           // prefetch next slab (DMA)
      __builtin_amdgcn_s_wait_asynccnt(4);    // this slab's 4 ops are done
    } else {
      __builtin_amdgcn_s_wait_asynccnt(0);
    }
    __syncthreads();                          // everyone's slab visible

    v16bf af[2], bfr[4];
#pragma unroll
    for (int rf = 0; rf < 2; ++rf)
      af[rf] = lds_frag(lA[buf], wm * 32 + rf * 16, lane);
#pragma unroll
    for (int cf = 0; cf < 4; ++cf)
      bfr[cf] = lds_frag(lW[buf], wn * 64 + cf * 16, lane);
#pragma unroll
    for (int rf = 0; rf < 2; ++rf)
#pragma unroll
      for (int cf = 0; cf < 4; ++cf)
        acc[rf][cf] = wmma_bf16(af[rf], bfr[cf], acc[rf][cf]);

    __syncthreads();                          // done reading before overwrite
  }
#else
  // Fallback: direct global fragment loads (L2-resident working set).
#pragma unroll 1
  for (int kk = 0; kk < DIM; kk += 32) {
    if (kk + 32 < DIM) {
      __builtin_prefetch(A + (size_t)(m0 + (lane & 15)) * DIM + kk + 32, 0, 0);
    }
    v16bf af[2], bfr[4];
#pragma unroll
    for (int rf = 0; rf < 2; ++rf)
      af[rf] = load_frag(A + (size_t)(m0 + rf * 16) * DIM + kk, DIM, lane);
#pragma unroll
    for (int cf = 0; cf < 4; ++cf)
      bfr[cf] = load_frag(W + (size_t)(n0 + cf * 16) * DIM + kk, DIM, lane);
#pragma unroll
    for (int rf = 0; rf < 2; ++rf)
#pragma unroll
      for (int cf = 0; cf < 4; ++cf)
        acc[rf][cf] = wmma_bf16(af[rf], bfr[cf], acc[rf][cf]);
  }
#endif

  // Epilogue. C-layout: VGPR r holds row (r + 8*hi), column = lo.
#pragma unroll
  for (int rf = 0; rf < 2; ++rf) {
#pragma unroll
    for (int cf = 0; cf < 4; ++cf) {
      const int rbase = m0 + rf * 16 + 8 * hi;
      const int col = n0 + cf * 16 + lo;
      if constexpr (MODE == 2) {
        float* out = (float*)outp;
        const float bv = bias[col];
#pragma unroll
        for (int r = 0; r < 8; ++r) {
          const int row = rbase + r;
          out[(size_t)row * DIM + col] = acc[rf][cf][r] * scale + bv;
        }
      } else if constexpr (MODE == 0) {
        bf16_t* out = (bf16_t*)outp;
        const int h = col >> 6, dd = col & 63;
#pragma unroll
        for (int r = 0; r < 8; ++r) {
          const int row = rbase + r;
          const int b = row >> 10, n = row & (SEQ - 1);
          out[((size_t)(b * HEADS + h) * SEQ + n) * HDIM + dd] =
              (bf16_t)(acc[rf][cf][r] * scale);
        }
      } else {  // MODE == 1 : transposed store [B,H,64,N], 8 consecutive n
        bf16_t* out = (bf16_t*)outp;
        const int h = col >> 6, dd = col & 63;
        const int b = rbase >> 10, nn0 = rbase & (SEQ - 1);
        v8bf pv;
#pragma unroll
        for (int r = 0; r < 8; ++r) pv[r] = (bf16_t)(acc[rf][cf][r] * scale);
        *reinterpret_cast<v8bf*>(
            out + ((size_t)(b * HEADS + h) * HDIM + dd) * SEQ + nn0) = pv;
      }
    }
  }
}

// ---------------------------------------------------------------------------
// Flash attention per (b,h). Block = 256 threads = 8 waves; each wave owns 32
// queries (full d=64 accumulated in registers) and streams 64-key tiles.
//   S = Q K^T (WMMA), mask, online softmax (shfl_xor row reductions over the
//   16-lane halves of the wave32 C-layout), P re-staged through wave-private
//   LDS (stride 72 bf16 = 144B: conflict-free, 16B aligned) to A-fragment
//   form, then O += P V via WMMA against V^T [B,H,64,N].
// Output: bf16 token-major [B*N, 1024] ready to feed the projection GEMM.
// ---------------------------------------------------------------------------
__global__ __launch_bounds__(256) void attn_kernel(
    const bf16_t* __restrict__ Q, const bf16_t* __restrict__ K,
    const bf16_t* __restrict__ VT, const unsigned char* __restrict__ mask,
    bf16_t* __restrict__ out) {
  __shared__ __align__(16) bf16_t plds[8][32][72];

  const int lane = threadIdx.x & 31;
  const int wave = threadIdx.x >> 5;
  const int lo = lane & 15, hi = lane >> 4;
  const int bh = blockIdx.y;
  const int b = bh >> 4, h = bh & 15;
  const int q0 = blockIdx.x * 256 + wave * 32;

  const bf16_t* Qb = Q + (size_t)bh * SEQ * HDIM;
  const bf16_t* Kb = K + (size_t)bh * SEQ * HDIM;
  const bf16_t* Vb = VT + (size_t)bh * HDIM * SEQ;
  const unsigned char* mb = mask + (size_t)b * (SEQ - 1);

  // Q fragments: 2 row-frags x 2 K-steps (d = 64), resident in VGPRs.
  v16bf qf[2][2];
#pragma unroll
  for (int rf = 0; rf < 2; ++rf)
#pragma unroll
    for (int ks = 0; ks < 2; ++ks)
      qf[rf][ks] =
          load_frag(Qb + (size_t)(q0 + rf * 16) * HDIM + ks * 32, HDIM, lane);

  v8f o[2][4] = {};
  float mrow[2][8], lrow[2][8];
#pragma unroll
  for (int rf = 0; rf < 2; ++rf)
#pragma unroll
    for (int r = 0; r < 8; ++r) {
      mrow[rf][r] = -1e30f;
      lrow[rf][r] = 0.0f;
    }

#pragma unroll 1
  for (int k0 = 0; k0 < SEQ; k0 += 64) {
    // ---- S = Q K^T on this 32x64 tile
    v8f s[2][4] = {};
#pragma unroll
    for (int cf = 0; cf < 4; ++cf) {
#pragma unroll
      for (int ks = 0; ks < 2; ++ks) {
        v16bf kf = load_frag(Kb + (size_t)(k0 + cf * 16) * HDIM + ks * 32,
                             HDIM, lane);
#pragma unroll
        for (int rf = 0; rf < 2; ++rf)
          s[rf][cf] = wmma_bf16(qf[rf][ks], kf, s[rf][cf]);
      }
    }

    // ---- mask: column j valid if j==0 (CLS) or mask[b, j-1]
#pragma unroll
    for (int cf = 0; cf < 4; ++cf) {
      const int j = k0 + cf * 16 + lo;
      const bool valid = (j == 0) ? true : (mb[j - 1] != 0);
      if (!valid) {
#pragma unroll
        for (int rf = 0; rf < 2; ++rf)
#pragma unroll
          for (int r = 0; r < 8; ++r) s[rf][cf][r] = -1e30f;
      }
    }

    // ---- online softmax: row = (rf, r, hi); columns live across lane half
#pragma unroll
    for (int rf = 0; rf < 2; ++rf) {
#pragma unroll
      for (int r = 0; r < 8; ++r) {
        float mx = fmaxf(fmaxf(s[rf][0][r], s[rf][1][r]),
                         fmaxf(s[rf][2][r], s[rf][3][r]));
#pragma unroll
        for (int off = 8; off > 0; off >>= 1)
          mx = fmaxf(mx, __shfl_xor(mx, off, 16));
        const float mold = mrow[rf][r];
        const float mnew = fmaxf(mold, mx);
        const float corr = __expf(mold - mnew);
        float rs = 0.0f;
#pragma unroll
        for (int cf = 0; cf < 4; ++cf) {
          const float p = __expf(s[rf][cf][r] - mnew);
          s[rf][cf][r] = p;
          rs += p;
        }
#pragma unroll
        for (int off = 8; off > 0; off >>= 1) rs += __shfl_xor(rs, off, 16);
        mrow[rf][r] = mnew;
        lrow[rf][r] = lrow[rf][r] * corr + rs;
#pragma unroll
        for (int df = 0; df < 4; ++df) o[rf][df][r] *= corr;
        // stage P into wave-private LDS tile (C-layout -> row-major)
        const int lr = rf * 16 + r + 8 * hi;
#pragma unroll
        for (int cf = 0; cf < 4; ++cf)
          plds[wave][lr][cf * 16 + lo] = (bf16_t)s[rf][cf][r];
      }
    }

    // ---- O += P V   (P re-read as A-fragments; V^T rows are d-columns)
#pragma unroll
    for (int ks = 0; ks < 2; ++ks) {
      v16bf pf[2];
#pragma unroll
      for (int rf = 0; rf < 2; ++rf)
        pf[rf] = load_frag(&plds[wave][rf * 16][ks * 32], 72, lane);
#pragma unroll
      for (int df = 0; df < 4; ++df) {
        v16bf vf = load_frag(Vb + (size_t)(df * 16) * SEQ + k0 + ks * 32,
                             SEQ, lane);
#pragma unroll
        for (int rf = 0; rf < 2; ++rf)
          o[rf][df] = wmma_bf16(pf[rf], vf, o[rf][df]);
      }
    }
  }

  // ---- normalize and store token-major [B*N, H*64]
#pragma unroll
  for (int rf = 0; rf < 2; ++rf) {
#pragma unroll
    for (int df = 0; df < 4; ++df) {
      const int col = h * HDIM + df * 16 + lo;
#pragma unroll
      for (int r = 0; r < 8; ++r) {
        const int n = q0 + rf * 16 + r + 8 * hi;
        out[((size_t)(b * SEQ + n)) * DIM + col] =
            (bf16_t)(o[rf][df][r] / lrow[rf][r]);
      }
    }
  }
}

// ---------------------------------------------------------------------------
// Host-side orchestration
// ---------------------------------------------------------------------------
extern "C" void kernel_launch(void* const* d_in, const int* in_sizes, int n_in,
                              void* d_out, int out_size, void* d_ws,
                              size_t ws_size, hipStream_t stream) {
  (void)in_sizes; (void)n_in; (void)out_size; (void)ws_size;

  const float* x = (const float*)d_in[0];
  const unsigned char* mask = (const unsigned char*)d_in[1];  // bool [B, N-1]
  const float* wq = (const float*)d_in[2];
  const float* wk = (const float*)d_in[3];
  const float* wv = (const float*)d_in[4];
  const float* wproj = (const float*)d_in[5];
  const float* bproj = (const float*)d_in[6];
  float* out = (float*)d_out;

  const int XN = BATCH * SEQ * DIM;  // 8388608
  const int WN = DIM * DIM;          // 1048576

  char* ws = (char*)d_ws;
  const size_t MB = 1024ull * 1024ull;
  bf16_t* xb   = (bf16_t*)(ws + 0 * MB);    // 16 MB
  bf16_t* wqb  = (bf16_t*)(ws + 16 * MB);   // 2 MB
  bf16_t* wkb  = (bf16_t*)(ws + 18 * MB);   // 2 MB
  bf16_t* wvb  = (bf16_t*)(ws + 20 * MB);   // 2 MB
  bf16_t* wpb  = (bf16_t*)(ws + 22 * MB);   // 2 MB
  bf16_t* qh   = (bf16_t*)(ws + 24 * MB);   // 16 MB  [B,H,N,64]  (pre-scaled)
  bf16_t* kh   = (bf16_t*)(ws + 40 * MB);   // 16 MB  [B,H,N,64]
  bf16_t* vT   = (bf16_t*)(ws + 56 * MB);   // 16 MB  [B,H,64,N]
  bf16_t* att  = (bf16_t*)(ws + 72 * MB);   // 16 MB  [B*N, 1024]

  // 1) fp32 -> bf16
  cvt_f32_to_bf16<<<XN / 256, 256, 0, stream>>>(x, xb, XN);
  cvt_f32_to_bf16<<<WN / 256, 256, 0, stream>>>(wq, wqb, WN);
  cvt_f32_to_bf16<<<WN / 256, 256, 0, stream>>>(wk, wkb, WN);
  cvt_f32_to_bf16<<<WN / 256, 256, 0, stream>>>(wv, wvb, WN);
  cvt_f32_to_bf16<<<WN / 256, 256, 0, stream>>>(wproj, wpb, WN);

  // 2) QKV projections (M=8192, N=1024, K=1024)
  const dim3 ggrid(DIM / 128, (BATCH * SEQ) / 128);
  const float scale = 0.125f;  // (d=64)^-0.5 folded into Q
  gemm_xwT<0><<<ggrid, 256, 0, stream>>>(xb, wqb, nullptr, scale, (void*)qh);
  gemm_xwT<0><<<ggrid, 256, 0, stream>>>(xb, wkb, nullptr, 1.0f, (void*)kh);
  gemm_xwT<1><<<ggrid, 256, 0, stream>>>(xb, wvb, nullptr, 1.0f, (void*)vT);

  // 3) flash attention: grid (query-tiles, B*H)
  attn_kernel<<<dim3(SEQ / 256, BATCH * HEADS), 256, 0, stream>>>(
      qh, kh, vT, mask, att);

  // 4) output projection + bias -> fp32
  gemm_xwT<2><<<ggrid, 256, 0, stream>>>(att, wpb, bproj, 1.0f, (void*)out);
}
